// MHConvAttention_6133213299150
// MI455X (gfx1250) — matrix operand order repaired and checked by reading, not compiled, not run on hardware
//
#include <hip/hip_runtime.h>

// ---------------------------------------------------------------------------
// MHConvAttention for MI455X (gfx1250): WMMA bf16 GEMM pipeline
//   - both 1x1 convs and the content-lambda batched GEMM on v_wmma bf16/f32
//   - double-buffered LDS staging, one barrier per k-step
//   - A-tile staging via GLOBAL_LOAD_ASYNC_TO_LDS_B128 when the builtin exists
// ---------------------------------------------------------------------------

#define B_   16
#define C_   128
#define H_   64
#define W_   64
#define N_   4096          // H*W
#define NH_  8
#define HD_  16
#define OUT_ 512

typedef __bf16 bf16x16 __attribute__((ext_vector_type(16)));
typedef float  floatx8 __attribute__((ext_vector_type(8)));
typedef unsigned int u32x4 __attribute__((ext_vector_type(4)));
typedef int v4i_ __attribute__((ext_vector_type(4)));

union BF16Frag {
    bf16x16 v;
    u32x4   q[2];
    unsigned short u[16];
};

__device__ __forceinline__ unsigned short f2bf(float f) {
    unsigned int x = __float_as_uint(f);
    unsigned int r = x + 0x7FFFu + ((x >> 16) & 1u);   // RNE
    return (unsigned short)(r >> 16);
}
__device__ __forceinline__ float bf2f(unsigned short u) {
    return __uint_as_float(((unsigned int)u) << 16);
}

// ---- CDNA5 async global->LDS path (guarded; falls back to plain copies) ----
#if defined(__AMDGCN__) && defined(__gfx1250__) && \
    __has_builtin(__builtin_amdgcn_global_load_async_to_lds_b128)
#define HAVE_ASYNC_LDS 1
#else
#define HAVE_ASYNC_LDS 0
#endif

#if HAVE_ASYNC_LDS
__device__ __forceinline__ void async16(const unsigned short* g, unsigned short* l) {
    // prototype (from compiler diagnostic): param0 is v4i* (global src),
    // param1 LDS dst, then imm offset, imm cpol
    __builtin_amdgcn_global_load_async_to_lds_b128(
        (v4i_*)g, (v4i_*)l, 0, 0);
}
__device__ __forceinline__ void async_wait_all() {
#if __has_builtin(__builtin_amdgcn_s_wait_asynccnt)
    __builtin_amdgcn_s_wait_asynccnt(0);
#else
    asm volatile("s_wait_asynccnt 0x0" ::: "memory");
#endif
}
#else
__device__ __forceinline__ void async_wait_all() {}
#endif

// ---------------------------------------------------------------------------
// fp32 -> bf16 weight conversion
// ---------------------------------------------------------------------------
__global__ __launch_bounds__(256) void cvt_kernel(const float* __restrict__ in,
                                                  unsigned short* __restrict__ out,
                                                  int n) {
    int i = blockIdx.x * 256 + threadIdx.x;
    if (i < n) out[i] = f2bf(in[i]);
}

// ---------------------------------------------------------------------------
// Global average pool over H*W per (b,c)
// ---------------------------------------------------------------------------
__global__ __launch_bounds__(256) void pool_kernel(const float* __restrict__ src,
                                                   float* __restrict__ pool) {
    int bc = blockIdx.x;
    const float* x = src + (size_t)bc * N_;
    int t = threadIdx.x;
    float s = 0.f;
    for (int j = 0; j < 16; ++j) s += x[t + j * 256];
    __shared__ float red[256];
    red[t] = s; __syncthreads();
    for (int k = 128; k > 0; k >>= 1) {
        if (t < k) red[t] += red[t + k];
        __syncthreads();
    }
    if (t == 0) pool[bc] = red[0] * (1.0f / (float)N_);
}

// ---------------------------------------------------------------------------
// ECA: 1D conv (width 3, zero pad) + sigmoid
// ---------------------------------------------------------------------------
__global__ __launch_bounds__(256) void ca_kernel(const float* __restrict__ pool,
                                                 const float* __restrict__ w3,
                                                 float* __restrict__ ca) {
    int idx = blockIdx.x * 256 + threadIdx.x;
    if (idx >= B_ * C_) return;
    int c = idx & (C_ - 1);
    float l = (c > 0)      ? pool[idx - 1] : 0.f;
    float m = pool[idx];
    float r = (c < C_ - 1) ? pool[idx + 1] : 0.f;
    float v = w3[0] * l + w3[1] * m + w3[2] * r;
    ca[idx] = 1.0f / (1.0f + __expf(-v));
}

// ---------------------------------------------------------------------------
// CPE: depthwise 3x3 + residual, output bf16
// ---------------------------------------------------------------------------
__global__ __launch_bounds__(256) void cpe_kernel(const float* __restrict__ src,
                                                  const float* __restrict__ cpe_w,
                                                  unsigned short* __restrict__ s_bf) {
    int bc = blockIdx.x;                 // b*C + c
    int c = bc & (C_ - 1);
    const float* x = src + (size_t)bc * N_;
    __shared__ float w[9];
    if (threadIdx.x < 9) w[threadIdx.x] = cpe_w[c * 9 + threadIdx.x];
    __syncthreads();
    for (int j = 0; j < 16; ++j) {
        int p = threadIdx.x + j * 256;
        int pr = p >> 6, pc = p & 63;
        float acc = x[p];                // residual
        #pragma unroll
        for (int dy = -1; dy <= 1; ++dy) {
            int r = pr + dy;
            if (r < 0 || r >= H_) continue;
            #pragma unroll
            for (int dx = -1; dx <= 1; ++dx) {
                int cc = pc + dx;
                if (cc < 0 || cc >= W_) continue;
                acc += w[(dy + 1) * 3 + (dx + 1)] * x[r * W_ + cc];
            }
        }
        s_bf[(size_t)bc * N_ + p] = f2bf(acc);
    }
}

// ---------------------------------------------------------------------------
// WMMA bf16 GEMM:  Out[z, M, n] = sum_k A[M,k] * Bmat[z, k, n]
//   A: [MTOT x KTOT] bf16 row-major (shared across batch)
//   Bmat: per-batch [KTOT x N_] bf16 rows
//   block = 256 threads = 8 waves; block tile M=128, N=32
//   wave w owns rows [w*16, w*16+16); computes two 16x16 tiles (shared A frag)
//   double-buffered LDS, single __syncthreads per k-step
// ---------------------------------------------------------------------------
template <int KTOT, int OUT_IS_BF16>
__global__ __launch_bounds__(256) void wmma_gemm_kernel(
        const unsigned short* __restrict__ A,
        const unsigned short* __restrict__ Bmat,
        unsigned short* __restrict__ OutBf,
        float* __restrict__ OutF,
        int BbatchStride, int OutBatchStride) {
    // Padded pitch 40 ushorts (80 B) keeps 16-byte alignment & spreads banks.
    __shared__ __align__(16) unsigned short a_s[2][128 * 40];
    __shared__ __align__(16) unsigned short b_s[2][32 * 40];   // transposed: [n][k]

    const int t    = threadIdx.x;
    const int lane = t & 31;
    const int wave = t >> 5;          // m-tile index within block
    const int n0   = blockIdx.x * 32;
    const int m0   = blockIdx.y * 128;
    const int z    = blockIdx.z;

    const unsigned short* Bb = Bmat + (size_t)z * BbatchStride;

    floatx8 acc0, acc1;
    #pragma unroll
    for (int r = 0; r < 8; ++r) { acc0[r] = 0.f; acc1[r] = 0.f; }

    const int ar0 = t >> 2;           // 0..63  (A chunk row; second chunk +64)
    const int ac0 = (t & 3) * 8;      // 0,8,16,24
    const int bk  = t >> 3;           // 0..31
    const int bn4 = (t & 7) * 4;      // 0..28

    auto stage = [&](int p, int kc) {
        // --- A: 128x32, two 16B chunks per thread ---
        const unsigned short* s0 = A + (size_t)(m0 + ar0) * KTOT + kc + ac0;
        const unsigned short* s1 = A + (size_t)(m0 + 64 + ar0) * KTOT + kc + ac0;
        unsigned short* d0 = &a_s[p][ar0 * 40 + ac0];
        unsigned short* d1 = &a_s[p][(64 + ar0) * 40 + ac0];
#if HAVE_ASYNC_LDS
        async16(s0, d0);
        async16(s1, d1);
#else
        *(u32x4*)d0 = *(const u32x4*)s0;
        *(u32x4*)d1 = *(const u32x4*)s1;
#endif
        // --- B: 32x32 transposed into b_s[n][k] (2-byte granular) ---
        const unsigned int* sp32 =
            (const unsigned int*)(Bb + (size_t)(kc + bk) * N_ + n0 + bn4);
        unsigned int w0 = sp32[0], w1 = sp32[1];
        b_s[p][(bn4 + 0) * 40 + bk] = (unsigned short)(w0 & 0xFFFFu);
        b_s[p][(bn4 + 1) * 40 + bk] = (unsigned short)(w0 >> 16);
        b_s[p][(bn4 + 2) * 40 + bk] = (unsigned short)(w1 & 0xFFFFu);
        b_s[p][(bn4 + 3) * 40 + bk] = (unsigned short)(w1 >> 16);
    };

    auto compute = [&](int p) {
        BF16Frag af, bf0, bf1;
        // A fragment per CDNA5 16-bit A 16x32 layout
        int row = wave * 16 + (lane & 15);
        int kb  = (lane >> 4) * 8;
        af.q[0] = *(const u32x4*)&a_s[p][row * 40 + kb];
        af.q[1] = *(const u32x4*)&a_s[p][row * 40 + kb + 16];
        // B fragments per CDNA5 16-bit B 32x16 layout
        int col = lane & 15;
        int kb2 = (lane >> 4) * 8;
        bf0.q[0] = *(const u32x4*)&b_s[p][col * 40 + kb2];
        bf0.q[1] = *(const u32x4*)&b_s[p][col * 40 + kb2 + 16];
        bf1.q[0] = *(const u32x4*)&b_s[p][(16 + col) * 40 + kb2];
        bf1.q[1] = *(const u32x4*)&b_s[p][(16 + col) * 40 + kb2 + 16];
        acc0 = __builtin_amdgcn_wmma_f32_16x16x32_bf16(
                   false, af.v, false, bf0.v, (short)0, acc0, false, false);
        acc1 = __builtin_amdgcn_wmma_f32_16x16x32_bf16(
                   false, af.v, false, bf1.v, (short)0, acc1, false, false);
    };

    constexpr int NSTEP = KTOT / 32;
    stage(0, 0);
    #pragma unroll
    for (int s = 0; s < NSTEP; ++s) {
        const int p = s & 1;
        async_wait_all();             // drain ASYNCcnt for buffer p
        __syncthreads();              // staging of p visible to all waves
        if (s + 1 < NSTEP) stage(p ^ 1, (s + 1) * 32);
        compute(p);
    }

    // --- write both D tiles: VGPR r -> M = r + 8*(lane>=16), N = lane&15 ---
    const int colb = n0 + (lane & 15);
    #pragma unroll
    for (int r = 0; r < 8; ++r) {
        int M = m0 + wave * 16 + r + 8 * (lane >> 4);
        size_t off = (size_t)z * OutBatchStride + (size_t)M * N_ + colb;
        if (OUT_IS_BF16) {
            OutBf[off]      = f2bf(acc0[r]);
            OutBf[off + 16] = f2bf(acc1[r]);
        } else {
            OutF[off]      = acc0[r];
            OutF[off + 16] = acc1[r];
        }
    }
}

// ---------------------------------------------------------------------------
// Softmax over spatial N per k-row, in place on the bf16 k region of qkv
// ---------------------------------------------------------------------------
__global__ __launch_bounds__(256) void softmax_kernel(unsigned short* __restrict__ qkv) {
    int row = blockIdx.x;                // b*C + c
    int b = row >> 7, c = row & (C_ - 1);
    unsigned short* p = qkv + (size_t)b * 3 * C_ * N_ + (size_t)(C_ + c) * N_;
    int t = threadIdx.x;
    float vals[16];
    float mx = -3.0e38f;
    for (int j = 0; j < 16; ++j) {
        float f = bf2f(p[t + j * 256]);
        vals[j] = f;
        mx = fmaxf(mx, f);
    }
    __shared__ float red[256];
    red[t] = mx; __syncthreads();
    for (int s = 128; s > 0; s >>= 1) {
        if (t < s) red[t] = fmaxf(red[t], red[t + s]);
        __syncthreads();
    }
    mx = red[0]; __syncthreads();
    float sum = 0.f;
    for (int j = 0; j < 16; ++j) { float e = __expf(vals[j] - mx); vals[j] = e; sum += e; }
    red[t] = sum; __syncthreads();
    for (int s = 128; s > 0; s >>= 1) {
        if (t < s) red[t] += red[t + s];
        __syncthreads();
    }
    float inv = 1.0f / red[0];
    for (int j = 0; j < 16; ++j) p[t + j * 256] = f2bf(vals[j] * inv);
}

// ---------------------------------------------------------------------------
// content_lambda[bh,i,o] = sum_n kf[i,n] * v[o,n]    (16 x 4096 x 16, WMMA)
// 8 waves split K=4096; LDS reduce; wave 0 writes 16x16 f32
// ---------------------------------------------------------------------------
__global__ __launch_bounds__(256) void lambda_kernel(const unsigned short* __restrict__ qkv,
                                                     float* __restrict__ lam) {
    int bh = blockIdx.x;
    int b = bh >> 3, h = bh & (NH_ - 1);
    int t = threadIdx.x, lane = t & 31, wave = t >> 5;
    const unsigned short* kf = qkv + (size_t)b * 3 * C_ * N_ + (size_t)(C_ + h * HD_) * N_;
    const unsigned short* vv = qkv + (size_t)b * 3 * C_ * N_ + (size_t)(2 * C_ + h * HD_) * N_;

    floatx8 acc;
    #pragma unroll
    for (int r = 0; r < 8; ++r) acc[r] = 0.f;

    const unsigned short* arow = kf + (size_t)(lane & 15) * N_;  // A row i = lane&15
    const unsigned short* brow = vv + (size_t)(lane & 15) * N_;  // B col o = lane&15
    const int akb = (lane >> 4) * 8;
    const int bkb = (lane >> 4) * 8;

    for (int step = 0; step < 16; ++step) {
        int nb = wave * 512 + step * 32;
        BF16Frag af, bfr;
        af.q[0]  = *(const u32x4*)(arow + nb + akb);
        af.q[1]  = *(const u32x4*)(arow + nb + akb + 16);
        bfr.q[0] = *(const u32x4*)(brow + nb + bkb);
        bfr.q[1] = *(const u32x4*)(brow + nb + bkb + 16);
        acc = __builtin_amdgcn_wmma_f32_16x16x32_bf16(
                false, af.v, false, bfr.v, (short)0, acc, false, false);
    }

    __shared__ float red[8 * 32 * 8];
    #pragma unroll
    for (int r = 0; r < 8; ++r) red[(wave * 32 + lane) * 8 + r] = acc[r];
    __syncthreads();
    if (wave == 0) {
        #pragma unroll
        for (int r = 0; r < 8; ++r) {
            float s = 0.f;
            for (int w2 = 0; w2 < 8; ++w2) s += red[(w2 * 32 + lane) * 8 + r];
            int M = r + 8 * (lane >> 4);      // i
            int Ncol = lane & 15;             // o
            lam[(size_t)bh * 256 + M * 16 + Ncol] = s;
        }
    }
}

// ---------------------------------------------------------------------------
// Combine: content einsum + 5x5 depthwise (rel_pos) * q + ECA scaling
// block per (rowchunk of 4 rows, head, batch); writes cat bf16 (B, 2C, N)
// ---------------------------------------------------------------------------
__global__ __launch_bounds__(256) void combine_kernel(
        const unsigned short* __restrict__ qkv,
        const float* __restrict__ lam,
        const float* __restrict__ rel_pos,
        const float* __restrict__ src,
        const float* __restrict__ ca,
        unsigned short* __restrict__ cat) {
    const int rc = blockIdx.x;           // 0..15 -> rows r0 = rc*4
    const int h  = blockIdx.y;
    const int b  = blockIdx.z;
    const int r0 = rc * 4;
    const int bh = b * NH_ + h;
    const int t  = threadIdx.x;

    __shared__ float lam_s[256];
    __shared__ float relw_s[400];
    __shared__ unsigned short q_s[16 * 256];   // [i][px]
    __shared__ unsigned short v_s[16 * 512];   // [o][8 halo rows][64]

    lam_s[t] = lam[(size_t)bh * 256 + t];
    relw_s[t] = rel_pos[t];
    if (t < 144) relw_s[256 + t] = rel_pos[256 + t];

    const unsigned short* qb = qkv + (size_t)b * 3 * C_ * N_ + (size_t)(h * HD_) * N_;
    const unsigned short* vb = qkv + (size_t)b * 3 * C_ * N_ + (size_t)(2 * C_ + h * HD_) * N_;

    for (int j = 0; j < 16; ++j) {           // q tile 16ch x 256px
        int e = t + j * 256;
        int ch = e >> 8, px = e & 255;
        q_s[ch * 256 + px] = qb[(size_t)ch * N_ + r0 * W_ + px];
    }
    for (int j = 0; j < 32; ++j) {           // v halo 16ch x 8rows x 64
        int e = t + j * 256;
        int ch = e >> 9, rr = (e >> 6) & 7, cc = e & 63;
        int grow = r0 + rr - 2;
        unsigned short val = 0;
        if (grow >= 0 && grow < H_) val = vb[(size_t)ch * N_ + grow * W_ + cc];
        v_s[ch * 512 + rr * 64 + cc] = val;
    }
    __syncthreads();

    const float scaling = 0.25f;             // HD^-0.5
    for (int j = 0; j < 16; ++j) {
        int e = t + j * 256;
        int o = e >> 8, px = e & 255;
        int pr = px >> 6, pc = px & 63;

        float content = 0.f;
        #pragma unroll
        for (int i = 0; i < 16; ++i)
            content += bf2f(q_s[i * 256 + px]) * lam_s[i * 16 + o];
        content *= scaling;

        float pl = 0.f;
        #pragma unroll
        for (int dy = 0; dy < 5; ++dy) {
            int hr = pr + dy;                 // halo row index
            #pragma unroll
            for (int dx = 0; dx < 5; ++dx) {
                int c2 = pc + dx - 2;
                if (c2 >= 0 && c2 < W_)
                    pl += relw_s[o * 25 + dy * 5 + dx] * bf2f(v_s[o * 512 + hr * 64 + c2]);
            }
        }
        float qv = bf2f(q_s[o * 256 + px]);
        float r1 = content + qv * pl;

        int cg = h * HD_ + o;
        size_t pix = (size_t)(r0 + pr) * W_ + pc;
        float sv = src[(size_t)b * C_ * N_ + (size_t)cg * N_ + pix];
        float r2 = sv * ca[b * C_ + cg];

        cat[(size_t)b * 2 * C_ * N_ + (size_t)cg * N_ + pix]        = f2bf(r1);
        cat[(size_t)b * 2 * C_ * N_ + (size_t)(C_ + cg) * N_ + pix] = f2bf(r2);
    }
}

// ---------------------------------------------------------------------------
// Launch
// ---------------------------------------------------------------------------
extern "C" void kernel_launch(void* const* d_in, const int* in_sizes, int n_in,
                              void* d_out, int out_size, void* d_ws, size_t ws_size,
                              hipStream_t stream) {
    const float* src      = (const float*)d_in[0];
    const float* cpe_w    = (const float*)d_in[1];
    const float* qkv_w    = (const float*)d_in[2];
    const float* rel_pos  = (const float*)d_in[3];
    const float* conv1d_w = (const float*)d_in[4];
    const float* out_w    = (const float*)d_in[5];
    float* out = (float*)d_out;

    char* ws = (char*)d_ws;
    size_t off = 0;
    auto alloc = [&](size_t bytes) -> void* {
        void* p = ws + off;
        off += (bytes + 255) & ~(size_t)255;
        return p;
    };
    unsigned short* s_bf  = (unsigned short*)alloc((size_t)B_ * C_ * N_ * 2);        // 16.8 MB
    unsigned short* qkv   = (unsigned short*)alloc((size_t)B_ * 3 * C_ * N_ * 2);    // 50.3 MB
    unsigned short* cat   = (unsigned short*)alloc((size_t)B_ * 2 * C_ * N_ * 2);    // 33.6 MB
    unsigned short* wqkv  = (unsigned short*)alloc((size_t)3 * C_ * C_ * 2);
    unsigned short* wout  = (unsigned short*)alloc((size_t)OUT_ * 2 * C_ * 2);
    float* lam  = (float*)alloc((size_t)B_ * NH_ * HD_ * HD_ * 4);
    float* pool = (float*)alloc((size_t)B_ * C_ * 4);
    float* cav  = (float*)alloc((size_t)B_ * C_ * 4);
    (void)ws_size; (void)in_sizes; (void)n_in; (void)out_size;

    cvt_kernel<<<(3 * C_ * C_ + 255) / 256, 256, 0, stream>>>(qkv_w, wqkv, 3 * C_ * C_);
    cvt_kernel<<<(OUT_ * 2 * C_ + 255) / 256, 256, 0, stream>>>(out_w, wout, OUT_ * 2 * C_);
    pool_kernel<<<B_ * C_, 256, 0, stream>>>(src, pool);
    ca_kernel<<<(B_ * C_ + 255) / 256, 256, 0, stream>>>(pool, conv1d_w, cav);
    cpe_kernel<<<B_ * C_, 256, 0, stream>>>(src, cpe_w, s_bf);

    // QKV GEMM: (384 x 128) x (128 x 4096) per batch -> bf16 q,k,v
    wmma_gemm_kernel<128, 1><<<dim3(N_ / 32, (3 * C_) / 128, B_), 256, 0, stream>>>(
        wqkv, s_bf, qkv, nullptr, C_ * N_, 3 * C_ * N_);

    softmax_kernel<<<B_ * C_, 256, 0, stream>>>(qkv);
    lambda_kernel<<<B_ * NH_, 256, 0, stream>>>(qkv, lam);
    combine_kernel<<<dim3(16, NH_, B_), 256, 0, stream>>>(qkv, lam, rel_pos, src, cav, cat);

    // Output GEMM: (512 x 256) x (256 x 4096) per batch -> f32 d_out
    wmma_gemm_kernel<256, 0><<<dim3(N_ / 32, OUT_ / 128, B_), 256, 0, stream>>>(
        wout, cat, nullptr, out, 2 * C_ * N_, OUT_ * N_);
}
